// MSAColumnAttention_1279900254328
// MI455X (gfx1250) — compile-verified, matrix operand order checked by reading, not run on hardware
//
#include <hip/hip_runtime.h>
#include <hip/hip_bf16.h>

// ---------------------------------------------------------------------------
// MSA column attention for MI455X (gfx1250, wave32, WMMA 16x16x32 f16)
// ---------------------------------------------------------------------------
#define S_DIM   256
#define I_DIM   384
#define CM      256
#define NHEADS  8
#define HDIM    32
#define TOTALT  256               // NHEADS * HDIM
#define NTOK    (S_DIM * I_DIM)   // 98304 tokens

typedef __attribute__((ext_vector_type(16))) _Float16 v16h;
typedef __attribute__((ext_vector_type(8)))  _Float16 v8h;
typedef __attribute__((ext_vector_type(8)))  float    v8f;

// ---- WMMA helpers ----------------------------------------------------------

__device__ __forceinline__ v8f wmma_f16(v16h a, v16h b, v8f c) {
    return __builtin_amdgcn_wmma_f32_16x16x32_f16(
        /*neg_a=*/false, a, /*neg_b=*/false, b,
        /*c_mod=*/(short)0, c, /*reuse_a=*/false, /*reuse_b=*/false);
}

// Load one 16x32 f16 A/B fragment from a row-major [rows x ld] f16 array.
// Layout per ISA 7.12.2: lane row = lane%16; lanes 0-15 hold K=0..7 (v0-3) and
// K=16..23 (v4-7); lanes 16-31 hold K=8..15 and K=24..31.
__device__ __forceinline__ v16h load_frag(const _Float16* p, int ld) {
    const int lane = threadIdx.x & 31;
    const int row  = lane & 15;
    const int kh   = (lane >> 4) << 3;          // 0 or 8
    const _Float16* r = p + row * ld;
    v8h lo = *(const v8h*)(r + kh);             // K = kh .. kh+7
    v8h hi = *(const v8h*)(r + kh + 16);        // K = 16+kh .. 16+kh+7
    v16h f;
#pragma unroll
    for (int i = 0; i < 8; ++i) { f[i] = lo[i]; f[i + 8] = hi[i]; }
    return f;
}

// ---------------------------------------------------------------------------
// Kernel 0: convert weights to f16.  Wcat = [Wq*inv_sqrt(D); Wk; Wv; Wg]
// (1024 x 256, row-major, rows = output channel, k contiguous).  Wo -> f16.
// ---------------------------------------------------------------------------
__global__ void k_convert_weights(const float* __restrict__ Wq,
                                  const float* __restrict__ Wk,
                                  const float* __restrict__ Wv,
                                  const float* __restrict__ Wg,
                                  const float* __restrict__ Wo,
                                  _Float16* __restrict__ Wcat,
                                  _Float16* __restrict__ Wo_h) {
    int idx = blockIdx.x * blockDim.x + threadIdx.x;      // 0 .. 5*65536-1
    const float qscale = 0.17677669529663687f;            // 1/sqrt(32)
    const int W = CM * TOTALT;                            // 65536
    if (idx < W)              Wcat[idx]         = (_Float16)(Wq[idx] * qscale);
    else if (idx < 2 * W)     Wcat[idx]         = (_Float16)Wk[idx - W];
    else if (idx < 3 * W)     Wcat[idx]         = (_Float16)Wv[idx - 2 * W];
    else if (idx < 4 * W)     Wcat[idx]         = (_Float16)Wg[idx - 3 * W];
    else if (idx < 5 * W)     Wo_h[idx - 4 * W] = (_Float16)Wo[idx - 4 * W];
}

// ---------------------------------------------------------------------------
// Kernel 1: LayerNorm over CM per token -> f16 activations.  One wave32/token.
// ---------------------------------------------------------------------------
__global__ void k_layernorm(const float* __restrict__ x,
                            const float* __restrict__ ln_w,
                            const float* __restrict__ ln_b,
                            _Float16* __restrict__ xh) {
    const int wave = (blockIdx.x * blockDim.x + threadIdx.x) >> 5;  // token id
    const int lane = threadIdx.x & 31;
    const float* row = x + (size_t)wave * CM;

    float v[8];
    float4 a = *(const float4*)(row + lane * 8);
    float4 b = *(const float4*)(row + lane * 8 + 4);
    v[0]=a.x; v[1]=a.y; v[2]=a.z; v[3]=a.w; v[4]=b.x; v[5]=b.y; v[6]=b.z; v[7]=b.w;

    float s = 0.f, ss = 0.f;
#pragma unroll
    for (int i = 0; i < 8; ++i) { s += v[i]; ss += v[i] * v[i]; }
#pragma unroll
    for (int m = 16; m >= 1; m >>= 1) {
        s  += __shfl_xor(s,  m, 32);
        ss += __shfl_xor(ss, m, 32);
    }
    const float mu  = s * (1.f / CM);
    const float var = ss * (1.f / CM) - mu * mu;
    const float inv = rsqrtf(var + 1e-5f);

    v8h o;
#pragma unroll
    for (int i = 0; i < 8; ++i) {
        const int c = lane * 8 + i;
        o[i] = (_Float16)((v[i] - mu) * inv * ln_w[c] + ln_b[c]);
    }
    *(v8h*)(xh + (size_t)wave * CM + lane * 8) = o;
}

// ---------------------------------------------------------------------------
// Kernel 2: fused Q/K/V/G projection.  Y[98304 x 1024] = xh @ Wcat^T.
// Block = 8 waves sharing one 16-row A strip staged in LDS (8 KB); each wave
// owns a 64-column group.  8 K-steps of wmma per accumulator.
// ---------------------------------------------------------------------------
__global__ void k_qkvg_gemm(const _Float16* __restrict__ xh,
                            const _Float16* __restrict__ Wcat,
                            const float*    __restrict__ bg,
                            _Float16* __restrict__ Qh, _Float16* __restrict__ Kh,
                            _Float16* __restrict__ Vh, _Float16* __restrict__ Gh) {
    __shared__ _Float16 Asm[16 * CM];            // 16-row x 256-k strip, 8 KB

    const int w    = threadIdx.x >> 5;           // 0..7
    const int lane = threadIdx.x & 31;
    const int mTile = blockIdx.x >> 1;           // 0 .. 6143
    const int half  = blockIdx.x & 1;            // n range halves
    const int m0 = mTile * 16;
    const int n0 = (half * 8 + w) * 64;          // 64 cols per wave

    // stage contiguous A strip: 256 threads x 16 halfs (32 B) each
    {
        const _Float16* src = xh + (size_t)m0 * CM + threadIdx.x * 16;
        v8h a0 = *(const v8h*)src;
        v8h a1 = *(const v8h*)(src + 8);
        *(v8h*)(Asm + threadIdx.x * 16)     = a0;
        *(v8h*)(Asm + threadIdx.x * 16 + 8) = a1;
    }
    __syncthreads();

    v8f zero = {};
    v8f acc[4] = {zero, zero, zero, zero};

#pragma unroll
    for (int k0 = 0; k0 < CM; k0 += 32) {
        v16h a = load_frag(Asm + k0, CM);        // ds_load path
#pragma unroll
        for (int j = 0; j < 4; ++j) {
            v16h b = load_frag(Wcat + (size_t)(n0 + j * 16) * CM + k0, CM);
            acc[j] = wmma_f16(a, b, acc[j]);
        }
    }

    const int nl    = lane & 15;
    const int mhalf = (lane >> 4) << 3;
#pragma unroll
    for (int j = 0; j < 4; ++j) {
        const int n     = n0 + j * 16 + nl;      // 0..1023
        const int which = n >> 8;                // uniform per (wave, j)
        const int t     = n & 255;
#pragma unroll
        for (int r = 0; r < 8; ++r) {
            const int m = m0 + mhalf + r;
            const size_t oidx = (size_t)m * TOTALT + t;
            const float val = acc[j][r];
            if (which == 0)      Qh[oidx] = (_Float16)val;   // pre-scaled via Wq
            else if (which == 1) Kh[oidx] = (_Float16)val;
            else if (which == 2) Vh[oidx] = (_Float16)val;
            else {
                const float g = val + bg[t];
                Gh[oidx] = (_Float16)(1.f / (1.f + __expf(-g)));
            }
        }
    }
}

// ---------------------------------------------------------------------------
// Kernel 3: column attention for one (i, h, s-half).  Block = 256 thr = 8 waves.
// K tile staged via GLOBAL_LOAD_ASYNC_TO_LDS_B128 (ASYNCcnt); V^T staged via
// manual transpose; gates prefetched (global_prefetch_b8).  Wave w handles
// rows s = sBase + w*16..+15: scores (16 wmma) -> shuffle softmax -> P restripe
// through per-wave LDS patch -> P@V (16 wmma) -> gate -> vals (f16).
// ---------------------------------------------------------------------------
__global__ void k_attention(const _Float16* __restrict__ Qh,
                            const _Float16* __restrict__ Kh,
                            const _Float16* __restrict__ Vh,
                            const _Float16* __restrict__ Gh,
                            const float*    __restrict__ mask,
                            _Float16* __restrict__ vals) {
    __shared__ _Float16 Kt[S_DIM * HDIM];        // K[t][d]        16 KB
    __shared__ _Float16 Vt[HDIM * S_DIM];        // V^T[d][t]      16 KB
    __shared__ _Float16 Pbuf[8][16 * 32];        // per-wave P      8 KB
    __shared__ float    mbias[S_DIM];            //                 1 KB

    const int sHalf = blockIdx.x;                // 0,1
    const int h     = blockIdx.y;                // 0..7
    const int i     = blockIdx.z;                // 0..383
    const int w     = threadIdx.x >> 5;
    const int lane  = threadIdx.x & 31;
    const int rowStride = I_DIM * TOTALT;        // 98304 f16 per s step

    // ---- stage K (async DMA to LDS), V^T (transpose), bias, gate prefetch --
    {
        const int t = threadIdx.x;               // 0..255 : one K/V row each
        const size_t base = ((size_t)(t * I_DIM + i)) * TOTALT + h * HDIM;

        // K row: 64 B per thread, per-lane async gather into LDS.
        // Low 32 bits of a generic LDS pointer are the LDS byte address
        // (ISA 10.2).
        {
            const unsigned lK = (unsigned)(uintptr_t)(&Kt[t * HDIM]);
            const unsigned long long gK =
                (unsigned long long)(uintptr_t)(Kh + base);
            asm volatile(
                "global_load_async_to_lds_b128 %0, %1, off\n\t"
                "global_load_async_to_lds_b128 %0, %1, off offset:16\n\t"
                "global_load_async_to_lds_b128 %0, %1, off offset:32\n\t"
                "global_load_async_to_lds_b128 %0, %1, off offset:48"
                :: "v"(lK), "v"(gK) : "memory");
        }

        // prefetch this row's gate values (needed only in the epilogue)
        __builtin_prefetch(Gh + base, 0, 1);

        // V^T transpose (scatter -> can't use async DMA)
        const _Float16* vrow = Vh + base;
#pragma unroll
        for (int d = 0; d < HDIM; ++d)
            Vt[d * S_DIM + t] = vrow[d];

        mbias[t] = (1.0f - mask[t * I_DIM + i]) * -1e9f;
    }
    asm volatile("s_wait_asynccnt 0" ::: "memory");   // own wave's async done
    __syncthreads();                                  // cross-wave visibility

    const int s0 = sHalf * 128 + w * 16;         // first row of this wave
    const int nl    = lane & 15;
    const int mhalf = (lane >> 4) << 3;

    // ---- scores: Q[16x32] x K^T -> 16 tiles of 16x16 ----------------------
    v16h qa = load_frag(Qh + ((size_t)(s0 * I_DIM + i)) * TOTALT + h * HDIM,
                        rowStride);
    v8f zero = {};
    v8f sc[16];
#pragma unroll
    for (int t = 0; t < 16; ++t) {
        v16h kb = load_frag(Kt + t * 16 * HDIM, HDIM);   // B: rows=t, k=d contig
        sc[t] = wmma_f16(qa, kb, zero);
    }

    // ---- softmax over the 256 columns of each of this wave's 16 rows ------
    float bias[16];
#pragma unroll
    for (int t = 0; t < 16; ++t) bias[t] = mbias[t * 16 + nl];

    float rinv[8];
#pragma unroll
    for (int r = 0; r < 8; ++r) {
        float mx = -1e30f;
#pragma unroll
        for (int t = 0; t < 16; ++t) {
            const float v = sc[t][r] + bias[t];
            sc[t][r] = v;
            mx = fmaxf(mx, v);
        }
#pragma unroll
        for (int m = 8; m >= 1; m >>= 1)          // stay within 16-lane half
            mx = fmaxf(mx, __shfl_xor(mx, m, 32));
        float sum = 0.f;
#pragma unroll
        for (int t = 0; t < 16; ++t) {
            const float e = __expf(sc[t][r] - mx);
            sc[t][r] = e;
            sum += e;
        }
#pragma unroll
        for (int m = 8; m >= 1; m >>= 1)
            sum += __shfl_xor(sum, m, 32);
        rinv[r] = 1.f / sum;
    }

    // ---- P @ V : restripe P through LDS, 8 K-chunks of 32 ------------------
    _Float16* pb = &Pbuf[w][0];                   // [16 rows][32 cols]
    v8f o0 = zero, o1 = zero;
#pragma unroll
    for (int kc = 0; kc < 8; ++kc) {
#pragma unroll
        for (int u = 0; u < 2; ++u) {
            const int t2 = kc * 2 + u;
#pragma unroll
            for (int r = 0; r < 8; ++r)
                pb[(mhalf + r) * 32 + u * 16 + nl] =
                    (_Float16)(sc[t2][r] * rinv[r]);
        }
        v16h pa = load_frag(pb, 32);
        v16h b0 = load_frag(Vt + 0  * S_DIM + kc * 32, S_DIM);  // d rows 0-15
        v16h b1 = load_frag(Vt + 16 * S_DIM + kc * 32, S_DIM);  // d rows 16-31
        o0 = wmma_f16(pa, b0, o0);
        o1 = wmma_f16(pa, b1, o1);
    }

    // ---- gate & store ------------------------------------------------------
#pragma unroll
    for (int dt = 0; dt < 2; ++dt) {
        const v8f& o = dt ? o1 : o0;
#pragma unroll
        for (int r = 0; r < 8; ++r) {
            const int s = s0 + mhalf + r;
            const int d = dt * 16 + nl;
            const size_t idx =
                ((size_t)(s * I_DIM + i)) * TOTALT + h * HDIM + d;
            const float g = (float)Gh[idx];
            vals[idx] = (_Float16)(g * o[r]);
        }
    }
}

// ---------------------------------------------------------------------------
// Kernel 4: output projection. out[98304 x 256] = vals @ Wo^T + bo  (f32 out)
// Block = 4 waves (128 thr) sharing one LDS-staged A strip; wave w owns 64 cols.
// ---------------------------------------------------------------------------
__global__ void k_out_proj(const _Float16* __restrict__ vals,
                           const _Float16* __restrict__ Wo_h,
                           const float*    __restrict__ bo,
                           float* __restrict__ out) {
    __shared__ _Float16 Asm[16 * TOTALT];        // 8 KB

    const int w    = threadIdx.x >> 5;           // 0..3
    const int lane = threadIdx.x & 31;
    const int m0 = blockIdx.x * 16;
    const int n0 = w * 64;

    // stage contiguous A strip: 128 threads x 32 halfs (64 B) each
    {
        const _Float16* src = vals + (size_t)m0 * TOTALT + threadIdx.x * 32;
        _Float16* dst = Asm + threadIdx.x * 32;
#pragma unroll
        for (int c = 0; c < 4; ++c)
            *(v8h*)(dst + c * 8) = *(const v8h*)(src + c * 8);
    }
    __syncthreads();

    v8f zero = {};
    v8f acc[4] = {zero, zero, zero, zero};

#pragma unroll
    for (int k0 = 0; k0 < TOTALT; k0 += 32) {
        v16h a = load_frag(Asm + k0, TOTALT);
#pragma unroll
        for (int j = 0; j < 4; ++j) {
            v16h b = load_frag(Wo_h + (size_t)(n0 + j * 16) * TOTALT + k0, TOTALT);
            acc[j] = wmma_f16(a, b, acc[j]);
        }
    }

    const int nl    = lane & 15;
    const int mhalf = (lane >> 4) << 3;
#pragma unroll
    for (int j = 0; j < 4; ++j) {
        const int n = n0 + j * 16 + nl;
        const float bias = bo[n];
#pragma unroll
        for (int r = 0; r < 8; ++r) {
            const int m = m0 + mhalf + r;
            out[(size_t)m * CM + n] = acc[j][r] + bias;
        }
    }
}

// ---------------------------------------------------------------------------
extern "C" void kernel_launch(void* const* d_in, const int* in_sizes, int n_in,
                              void* d_out, int out_size, void* d_ws, size_t ws_size,
                              hipStream_t stream) {
    (void)in_sizes; (void)n_in; (void)out_size; (void)ws_size;

    const float* msa  = (const float*)d_in[0];
    const float* mask = (const float*)d_in[1];
    const float* ln_w = (const float*)d_in[2];
    const float* ln_b = (const float*)d_in[3];
    const float* Wq   = (const float*)d_in[4];
    const float* Wk   = (const float*)d_in[5];
    const float* Wv   = (const float*)d_in[6];
    const float* Wg   = (const float*)d_in[7];
    const float* bg   = (const float*)d_in[8];
    const float* Wo   = (const float*)d_in[9];
    const float* bo   = (const float*)d_in[10];
    float* out = (float*)d_out;

    // workspace carve-up (f16 buffers), 256B aligned
    char* ws = (char*)d_ws;
    size_t off = 0;
    auto carve = [&](size_t bytes) -> char* {
        char* p = ws + off;
        off = (off + bytes + 255) & ~(size_t)255;
        return p;
    };
    const size_t actBytes = (size_t)NTOK * TOTALT * sizeof(_Float16); // 50.3 MB
    _Float16* xh    = (_Float16*)carve(actBytes);
    _Float16* Qh    = (_Float16*)carve(actBytes);
    _Float16* Kh    = (_Float16*)carve(actBytes);
    _Float16* Vh    = (_Float16*)carve(actBytes);
    _Float16* Gh    = (_Float16*)carve(actBytes);
    _Float16* valsh = (_Float16*)carve(actBytes);
    _Float16* Wcat  = (_Float16*)carve((size_t)4 * CM * TOTALT * sizeof(_Float16));
    _Float16* Wo_h  = (_Float16*)carve((size_t)CM * TOTALT * sizeof(_Float16));

    // 0) weights -> f16 (1/sqrt(D) folded into Wq)
    k_convert_weights<<<(5 * CM * TOTALT + 255) / 256, 256, 0, stream>>>(
        Wq, Wk, Wv, Wg, Wo, Wcat, Wo_h);

    // 1) LayerNorm: one wave per token, 8 waves/block
    k_layernorm<<<NTOK / 8, 256, 0, stream>>>(msa, ln_w, ln_b, xh);

    // 2) Q/K/V/G projection: (NTOK/16) mTiles x 2 column-halves
    k_qkvg_gemm<<<(NTOK / 16) * 2, 256, 0, stream>>>(
        xh, Wcat, bg, Qh, Kh, Vh, Gh);

    // 3) attention: (s-half, head, column)
    dim3 agrid(2, NHEADS, I_DIM);
    k_attention<<<agrid, 256, 0, stream>>>(Qh, Kh, Vh, Gh, mask, valsh);

    // 4) output projection: one block (4 waves) per 16-row strip
    k_out_proj<<<NTOK / 16, 128, 0, stream>>>(valsh, Wo_h, bo, out);
}
